// RRGCATbert_model_68384469286920
// MI455X (gfx1250) — matrix operand order repaired and checked by reading, not compiled, not run on hardware
//
#include <hip/hip_runtime.h>
#include <hip/hip_bf16.h>

// ---------------------------------------------------------------------------
// Types for CDNA5 WMMA (wave32)
// ---------------------------------------------------------------------------
typedef __attribute__((ext_vector_type(16))) __bf16          v16bf;
typedef __attribute__((ext_vector_type(8)))  float           v8f;
typedef __attribute__((ext_vector_type(8)))  unsigned short  su8;
typedef __attribute__((ext_vector_type(16))) unsigned short  v16u;
typedef __attribute__((ext_vector_type(4)))  unsigned int    vu4;

#define N_TOK 2048
#define KDIM  1024

static __device__ __forceinline__ unsigned short f2bf(float f) {
    unsigned u = __builtin_bit_cast(unsigned, f);
    unsigned r = (u + 0x7FFFu + ((u >> 16) & 1u)) >> 16;   // RNE
    return (unsigned short)r;
}

static __device__ __forceinline__ v8f wmma_bf16(v16bf a, v16bf b, v8f c) {
    return __builtin_amdgcn_wmma_f32_16x16x32_bf16(
        /*neg_a=*/false, a, /*neg_b=*/false, b,
        /*c_mod=*/(short)0, c, /*reuse_a=*/false, /*reuse_b=*/false);
}

// A-matrix 16x32 bf16 frag (ISA 7.12.2): lane holds row M=lane%16; K-chunks
// [base0 .. base0+7] and [base0+16 .. base0+23], base0 = (lane>=16 ? 8 : 0).
static __device__ __forceinline__ v16bf load_a_g(const unsigned short* p, int lane) {
    const su8* q = reinterpret_cast<const su8*>(p + ((lane & 16) ? 8 : 0));
    su8 c0 = q[0];           // halves +0..7
    su8 c1 = q[2];           // halves +16..23
    v16u r;
#pragma unroll
    for (int j = 0; j < 8; ++j) { r[j] = c0[j]; r[j + 8] = c1[j]; }
    return __builtin_bit_cast(v16bf, r);
}

// B-matrix 32x16 bf16 frag: lane holds col N=lane%16; K-chunks
// [base0 .. base0+7] and [base0+16 .. base0+23], base0 = (lane>=16 ? 8 : 0).
static __device__ __forceinline__ v16bf load_b_g(const unsigned short* p, int lane) {
    const su8* q = reinterpret_cast<const su8*>(p + ((lane & 16) ? 8 : 0));
    su8 c0 = q[0];
    su8 c1 = q[2];
    v16u r;
#pragma unroll
    for (int j = 0; j < 8; ++j) { r[j] = c0[j]; r[j + 8] = c1[j]; }
    return __builtin_bit_cast(v16bf, r);
}

// A-frag built from f32 probabilities living in LDS (convert to bf16 inline).
static __device__ __forceinline__ v16bf load_a_lds_f32(const float* p, int lane) {
    const float4* q = reinterpret_cast<const float4*>(p + ((lane & 16) ? 8 : 0));
    float4 c0 = q[0], c1 = q[1];   // floats +0..7
    float4 c2 = q[4], c3 = q[5];   // floats +16..23
    v16u r;
    r[0]  = f2bf(c0.x); r[1]  = f2bf(c0.y); r[2]  = f2bf(c0.z); r[3]  = f2bf(c0.w);
    r[4]  = f2bf(c1.x); r[5]  = f2bf(c1.y); r[6]  = f2bf(c1.z); r[7]  = f2bf(c1.w);
    r[8]  = f2bf(c2.x); r[9]  = f2bf(c2.y); r[10] = f2bf(c2.z); r[11] = f2bf(c2.w);
    r[12] = f2bf(c3.x); r[13] = f2bf(c3.y); r[14] = f2bf(c3.z); r[15] = f2bf(c3.w);
    return __builtin_bit_cast(v16bf, r);
}

static __device__ __forceinline__ float wred_max(float v) {
#pragma unroll
    for (int o = 16; o > 0; o >>= 1) v = fmaxf(v, __shfl_xor(v, o, 32));
    return v;
}
static __device__ __forceinline__ float wred_sum(float v) {
#pragma unroll
    for (int o = 16; o > 0; o >>= 1) v += __shfl_xor(v, o, 32);
    return v;
}

// ---------------------------------------------------------------------------
// Prep kernels
// ---------------------------------------------------------------------------
__global__ void cvt_f32_bf16_kernel(const float* __restrict__ in,
                                    unsigned short* __restrict__ out, int n) {
    int i = blockIdx.x * 256 + threadIdx.x;
    if (i < n) out[i] = f2bf(in[i]);
}

// W (1024 x Nc, f32)  ->  Wt (Nc x 1024, bf16)
__global__ void wtrans_kernel(const float* __restrict__ W,
                              unsigned short* __restrict__ Wt, int Nc) {
    __shared__ float tile[32][33];
    const int k0 = blockIdx.x * 32, n0 = blockIdx.y * 32;
    const int tx = threadIdx.x, ty = threadIdx.y;
#pragma unroll
    for (int i = 0; i < 32; i += 8)
        tile[ty + i][tx] = W[(size_t)(k0 + ty + i) * Nc + n0 + tx];
    __syncthreads();
#pragma unroll
    for (int i = 0; i < 32; i += 8)
        Wt[(size_t)(n0 + ty + i) * KDIM + k0 + tx] = f2bf(tile[tx][ty + i]);
}

// Mt[n][m] = (G[m][n] - 1) * 10000   (pre-transposed additive mask)
__global__ void mtrans_kernel(const float* __restrict__ G, float* __restrict__ Mt) {
    __shared__ float tile[32][33];
    const int m0 = blockIdx.x * 32, n0 = blockIdx.y * 32;
    const int tx = threadIdx.x, ty = threadIdx.y;
#pragma unroll
    for (int i = 0; i < 32; i += 8)
        tile[ty + i][tx] = G[(size_t)(m0 + ty + i) * N_TOK + n0 + tx];
    __syncthreads();
#pragma unroll
    for (int i = 0; i < 32; i += 8)
        Mt[(size_t)(n0 + ty + i) * N_TOK + m0 + tx] = (tile[tx][ty + i] - 1.0f) * 10000.0f;
}

// ---------------------------------------------------------------------------
// bf16 WMMA GEMM:  out(bf16) = A(2048x1024 bf16) @ Bt^T + bias
//   Bt is NC x K (pre-transposed weights).  STORET -> out is NC x 2048 (for V).
//   NC/STORET are compile-time: branchless epilogue, shift addressing, and a
//   single b128 store per tile in the transposed path.
//   WG = 8 waves, each wave = 64x32 tile, WG tile = 128x128.
// ---------------------------------------------------------------------------
template <int NC, bool STORET>
__global__ __launch_bounds__(256)
void gemm_qkv_kernel(const unsigned short* __restrict__ A,
                     const unsigned short* __restrict__ Bt,
                     const float* __restrict__ bias,
                     unsigned short* __restrict__ out) {
    const int tid  = threadIdx.x;
    const int wave = tid >> 5, lane = tid & 31;
    const int m16  = lane & 15, hi8 = (lane & 16) ? 8 : 0;
    const int wm = wave >> 2, wn = wave & 3;
    const int mbase = blockIdx.x * 128 + wm * 64;
    const int nbase = blockIdx.y * 128 + wn * 32;

    const unsigned short* arow[4];
#pragma unroll
    for (int mt = 0; mt < 4; ++mt) arow[mt] = A  + (size_t)(mbase + mt * 16 + m16) * KDIM;
    const unsigned short* brow[2];
#pragma unroll
    for (int nt = 0; nt < 2; ++nt) brow[nt] = Bt + (size_t)(nbase + nt * 16 + m16) * KDIM;

    v8f acc[4][2] = {};
    for (int kb = 0; kb < KDIM; kb += 32) {
        v16bf af[4];
#pragma unroll
        for (int mt = 0; mt < 4; ++mt) af[mt] = load_a_g(arow[mt] + kb, lane);
        v16bf bf2[2];
#pragma unroll
        for (int nt = 0; nt < 2; ++nt) bf2[nt] = load_b_g(brow[nt] + kb, lane);
#pragma unroll
        for (int mt = 0; mt < 4; ++mt)
#pragma unroll
            for (int nt = 0; nt < 2; ++nt)
                acc[mt][nt] = wmma_bf16(af[mt], bf2[nt], acc[mt][nt]);
    }

#pragma unroll
    for (int nt = 0; nt < 2; ++nt) {
        const int col = nbase + nt * 16 + m16;
        const float bv = bias[col];
#pragma unroll
        for (int mt = 0; mt < 4; ++mt) {
            if (STORET) {
                // 8 consecutive bf16 along the output row -> one b128 store
                su8 pk;
#pragma unroll
                for (int g = 0; g < 8; ++g) pk[g] = f2bf(acc[mt][nt][g] + bv);
                const int row0 = mbase + mt * 16 + hi8;
                *reinterpret_cast<vu4*>(out + (size_t)col * N_TOK + row0) =
                    __builtin_bit_cast(vu4, pk);
            } else {
#pragma unroll
                for (int g = 0; g < 8; ++g) {
                    const int row = mbase + mt * 16 + g + hi8;
                    out[(size_t)row * NC + col] = f2bf(acc[mt][nt][g] + bv);
                }
            }
        }
    }
}

// ---------------------------------------------------------------------------
// Fused attention: one WG = one head x 16 query rows.
//  Phase 1: S1 = scale*Q1 K1^T, S2 = scale*Q2 K2^T   (full 16x2048 rows in LDS)
//  Phase 2: in-LDS gather (n2c/c2n) + mask + softmax (unnormalized, keep 1/sum)
//  Phase 3: out = (P @ V) * (1/sum)  via WMMA, V pre-transposed in bf16.
//  LDS: 2 * 16 * 2048 * 4B = 256 KB dynamic.
// ---------------------------------------------------------------------------
__global__ __launch_bounds__(256)
void attn_kernel(const unsigned short* __restrict__ Q1,
                 const unsigned short* __restrict__ K1m,
                 const unsigned short* __restrict__ Q2,
                 const unsigned short* __restrict__ K2m,
                 const unsigned short* __restrict__ V1t,
                 const unsigned short* __restrict__ V2t,
                 const int*   __restrict__ n2c,
                 const int*   __restrict__ c2n,
                 const float* __restrict__ m1t,
                 const float* __restrict__ m2t,
                 float* __restrict__ out) {
    extern __shared__ float smem[];
    float* S1 = smem;                 // 16 x 2048
    float* S2 = smem + 16 * 2048;     // 16 x 2048
    __shared__ float rowinv1[16], rowinv2[16];

    const int h  = blockIdx.y;
    const int n0 = blockIdx.x * 16;
    const int tid  = threadIdx.x;
    const int wave = tid >> 5, lane = tid & 31;
    const int m16  = lane & 15, hi8 = (lane & 16) ? 8 : 0;
    const float scale = 0.08838834764831845f;  // 1/sqrt(128)

    // ---- Phase 1: scores -------------------------------------------------
    {
        const unsigned short* q1r = Q1 + (size_t)(n0 + m16) * KDIM + h * 128;
        const unsigned short* q2r = Q2 + (size_t)(n0 + m16) * KDIM + h * 128;
        v16bf a1f[4], a2f[4];
#pragma unroll
        for (int kk = 0; kk < 4; ++kk) {
            a1f[kk] = load_a_g(q1r + kk * 32, lane);
            a2f[kk] = load_a_g(q2r + kk * 32, lane);
        }
        for (int t = wave; t < 128; t += 8) {        // m-tiles across 8 waves
            const int mcol = t * 16;
            const unsigned short* k1r = K1m + (size_t)(mcol + m16) * KDIM + h * 128;
            const unsigned short* k2r = K2m + (size_t)(mcol + m16) * KDIM + h * 128;
            if (t + 8 < 128) {  // prefetch next tile's K rows (global_prefetch_b8)
                __builtin_prefetch(K1m + (size_t)(mcol + 128 + m16) * KDIM + h * 128, 0, 1);
                __builtin_prefetch(K2m + (size_t)(mcol + 128 + m16) * KDIM + h * 128, 0, 1);
            }
            v8f acc1 = {}, acc2 = {};
#pragma unroll
            for (int kk = 0; kk < 4; ++kk) {
                acc1 = wmma_bf16(a1f[kk], load_b_g(k1r + kk * 32, lane), acc1);
                acc2 = wmma_bf16(a2f[kk], load_b_g(k2r + kk * 32, lane), acc2);
            }
#pragma unroll
            for (int g = 0; g < 8; ++g) {
                const int r = g + hi8;
                S1[r * 2048 + mcol + m16] = acc1[g] * scale;
                S2[r * 2048 + mcol + m16] = acc2[g] * scale;
            }
        }
    }
    __syncthreads();

    // ---- Phase 2: gather + mask + softmax (rows are wave-private) --------
#pragma unroll 1
    for (int rr = 0; rr < 2; ++rr) {
        const int r = wave + rr * 8;
        const int n = n0 + r;
        const int*   n2cr = n2c + (size_t)n * 2048;
        const int*   c2nr = c2n + (size_t)n * 2048;
        const float* m1r  = m1t + (size_t)n * 2048;
        const float* m2r  = m2t + (size_t)n * 2048;
        float* s1r = S1 + r * 2048;
        float* s2r = S2 + r * 2048;

        float t2buf[64];
        float mx1 = -3.0e38f, mx2 = -3.0e38f;
        // t2 = s2 + gather(s1, n2c) + m2   (buffer in regs: s1 row mutates next)
#pragma unroll
        for (int i = 0; i < 64; ++i) {
            const int m = i * 32 + lane;
            const float v = s2r[m] + s1r[n2cr[m]] + m2r[m];
            t2buf[i] = v;
            mx2 = fmaxf(mx2, v);
        }
        // t1 = s1 + gather(s2, c2n) + m1   (in-place into s1 row)
#pragma unroll
        for (int i = 0; i < 64; ++i) {
            const int m = i * 32 + lane;
            const float u = s1r[m] + s2r[c2nr[m]] + m1r[m];
            s1r[m] = u;
            mx1 = fmaxf(mx1, u);
        }
        mx1 = wred_max(mx1);
        mx2 = wred_max(mx2);
        float sum1 = 0.f, sum2 = 0.f;
#pragma unroll
        for (int i = 0; i < 64; ++i) {
            const int m = i * 32 + lane;
            const float e1 = __expf(s1r[m] - mx1);
            s1r[m] = e1; sum1 += e1;
            const float e2 = __expf(t2buf[i] - mx2);
            s2r[m] = e2; sum2 += e2;
        }
        sum1 = wred_sum(sum1);
        sum2 = wred_sum(sum2);
        if (lane == 0) { rowinv1[r] = 1.0f / sum1; rowinv2[r] = 1.0f / sum2; }
    }
    __syncthreads();

    // ---- Phase 3: P @ V --------------------------------------------------
    {
        const int b     = wave >> 2;   // 0 = branch1, 1 = branch2
        const int ntile = wave & 3;    // 4 tiles of 16 over head dim 64
        const float*          P    = b ? S2 : S1;
        const unsigned short* Vt   = b ? V2t : V1t;
        const float*          rinv = b ? rowinv2 : rowinv1;
        const unsigned short* vr = Vt + (size_t)(h * 64 + ntile * 16 + m16) * N_TOK;
        const float*          pr = P + m16 * 2048;
        v8f acc = {};
        for (int kk = 0; kk < 64; ++kk) {
            v16bf a  = load_a_lds_f32(pr + kk * 32, lane);
            v16bf bb = load_b_g(vr + kk * 32, lane);
            acc = wmma_bf16(a, bb, acc);
        }
#pragma unroll
        for (int g = 0; g < 8; ++g) {
            const int r = g + hi8;
            out[(size_t)(n0 + r) * 1024 + b * 512 + h * 64 + ntile * 16 + m16]
                = acc[g] * rinv[r];
        }
    }
}

// ---------------------------------------------------------------------------
// Host launcher
// ---------------------------------------------------------------------------
extern "C" void kernel_launch(void* const* d_in, const int* in_sizes, int n_in,
                              void* d_out, int out_size, void* d_ws, size_t ws_size,
                              hipStream_t stream) {
    const float* x   = (const float*)d_in[0];
    const float* fst = (const float*)d_in[1];
    const float* sec = (const float*)d_in[2];
    const int*   n2c = (const int*)d_in[3];
    const int*   c2n = (const int*)d_in[4];
    const float* Wq1 = (const float*)d_in[5];  const float* bq1 = (const float*)d_in[6];
    const float* Wk1 = (const float*)d_in[7];  const float* bk1 = (const float*)d_in[8];
    const float* Wv1 = (const float*)d_in[9];  const float* bv1 = (const float*)d_in[10];
    const float* Wq2 = (const float*)d_in[11]; const float* bq2 = (const float*)d_in[12];
    const float* Wk2 = (const float*)d_in[13]; const float* bk2 = (const float*)d_in[14];
    const float* Wv2 = (const float*)d_in[15]; const float* bv2 = (const float*)d_in[16];
    float* out = (float*)d_out;

    char* ws = (char*)d_ws;
    size_t off = 0;
    auto alloc = [&](size_t bytes) -> void* {
        void* p = ws + off;
        off += (bytes + 255) & ~(size_t)255;
        return p;
    };
    unsigned short* xb   = (unsigned short*)alloc((size_t)N_TOK * KDIM * 2);
    unsigned short* wq1t = (unsigned short*)alloc((size_t)1024 * KDIM * 2);
    unsigned short* wk1t = (unsigned short*)alloc((size_t)1024 * KDIM * 2);
    unsigned short* wq2t = (unsigned short*)alloc((size_t)1024 * KDIM * 2);
    unsigned short* wk2t = (unsigned short*)alloc((size_t)1024 * KDIM * 2);
    unsigned short* wv1t = (unsigned short*)alloc((size_t)512 * KDIM * 2);
    unsigned short* wv2t = (unsigned short*)alloc((size_t)512 * KDIM * 2);
    unsigned short* q1   = (unsigned short*)alloc((size_t)N_TOK * 1024 * 2);
    unsigned short* k1   = (unsigned short*)alloc((size_t)N_TOK * 1024 * 2);
    unsigned short* q2   = (unsigned short*)alloc((size_t)N_TOK * 1024 * 2);
    unsigned short* k2   = (unsigned short*)alloc((size_t)N_TOK * 1024 * 2);
    unsigned short* v1t  = (unsigned short*)alloc((size_t)512 * N_TOK * 2);
    unsigned short* v2t  = (unsigned short*)alloc((size_t)512 * N_TOK * 2);
    float* m1t = (float*)alloc((size_t)N_TOK * N_TOK * 4);
    float* m2t = (float*)alloc((size_t)N_TOK * N_TOK * 4);
    (void)ws_size; (void)in_sizes; (void)n_in; (void)out_size;

    // 1) precision / layout prep
    {
        const int n = N_TOK * KDIM;
        cvt_f32_bf16_kernel<<<n / 256, 256, 0, stream>>>(x, xb, n);
        dim3 blk(32, 8);
        wtrans_kernel<<<dim3(32, 32), blk, 0, stream>>>(Wq1, wq1t, 1024);
        wtrans_kernel<<<dim3(32, 32), blk, 0, stream>>>(Wk1, wk1t, 1024);
        wtrans_kernel<<<dim3(32, 32), blk, 0, stream>>>(Wq2, wq2t, 1024);
        wtrans_kernel<<<dim3(32, 32), blk, 0, stream>>>(Wk2, wk2t, 1024);
        wtrans_kernel<<<dim3(32, 16), blk, 0, stream>>>(Wv1, wv1t, 512);
        wtrans_kernel<<<dim3(32, 16), blk, 0, stream>>>(Wv2, wv2t, 512);
        mtrans_kernel<<<dim3(64, 64), blk, 0, stream>>>(fst, m1t);
        mtrans_kernel<<<dim3(64, 64), blk, 0, stream>>>(sec, m2t);
    }

    // 2) QKV projections (bf16 WMMA)
    {
        dim3 g1024(N_TOK / 128, 1024 / 128);
        dim3 g512 (N_TOK / 128, 512 / 128);
        gemm_qkv_kernel<1024, false><<<g1024, 256, 0, stream>>>(xb, wq1t, bq1, q1);
        gemm_qkv_kernel<1024, false><<<g1024, 256, 0, stream>>>(xb, wk1t, bk1, k1);
        gemm_qkv_kernel<1024, false><<<g1024, 256, 0, stream>>>(xb, wq2t, bq2, q2);
        gemm_qkv_kernel<1024, false><<<g1024, 256, 0, stream>>>(xb, wk2t, bk2, k2);
        gemm_qkv_kernel<512,  true ><<<g512,  256, 0, stream>>>(xb, wv1t, bv1, v1t);
        gemm_qkv_kernel<512,  true ><<<g512,  256, 0, stream>>>(xb, wv2t, bv2, v2t);
    }

    // 3) fused dual-branch attention (256 KB dynamic LDS per WG)
    {
        const size_t lds = (size_t)2 * 16 * 2048 * sizeof(float);
        (void)hipFuncSetAttribute((const void*)attn_kernel,
                                  hipFuncAttributeMaxDynamicSharedMemorySize,
                                  (int)lds);
        attn_kernel<<<dim3(N_TOK / 16, 8), 256, lds, stream>>>(
            q1, k1, q2, k2, v1t, v2t, n2c, c2n, m1t, m2t, out);
    }
}
